// LocalWindowMultiHeadAttentionFast_8211977470052
// MI455X (gfx1250) — matrix operand order repaired and checked by reading, not compiled, not run on hardware
//
#include <hip/hip_runtime.h>
#include <hip/hip_bf16.h>

// ---- problem constants -----------------------------------------------------
#define L_   2048
#define B_   8
#define D_   512
#define H_   8
#define DH_  64
#define PAD_ 64
#define M_   (L_ * B_)      // 16384 rows for both GEMMs
#define NEGF (-1.0e30f)

// ---- vector types ----------------------------------------------------------
typedef __attribute__((ext_vector_type(16))) __bf16 v16bf;
typedef __attribute__((ext_vector_type(8)))  __bf16 v8bf;
typedef __attribute__((ext_vector_type(4)))  __bf16 v4bf;
typedef __attribute__((ext_vector_type(8)))  float  v8f;
typedef __attribute__((ext_vector_type(4)))  float  v4f;

__device__ __forceinline__ v16bf cat8(v8bf lo, v8bf hi) {
  return __builtin_shufflevector(lo, hi, 0,1,2,3,4,5,6,7,8,9,10,11,12,13,14,15);
}
__device__ __forceinline__ v4bf cvt4(v4f f) {
  v4bf r; r[0]=(__bf16)f[0]; r[1]=(__bf16)f[1]; r[2]=(__bf16)f[2]; r[3]=(__bf16)f[3];
  return r;
}
#define WMMA_BF16(A,B,C) \
  __builtin_amdgcn_wmma_f32_16x16x32_bf16(false,(A),false,(B),(short)0,(C),false,false)

// ---- CDNA5 async global->LDS copy (16 bytes / lane, ASYNCcnt tracked) ------
__device__ __forceinline__ void async_cp16(const __bf16* g, __bf16* l) {
  uint32_t loff = (uint32_t)(uintptr_t)l;          // flat LDS addr[31:0] = LDS byte addr
  uint64_t gaddr = (uint64_t)(uintptr_t)g;
  asm volatile("global_load_async_to_lds_b128 %0, %1, off"
               :: "v"(loff), "v"(gaddr) : "memory");
}
#define WAIT_ASYNC(n) asm volatile("s_wait_asynccnt " #n ::: "memory")

// ---- GEMM tiling -----------------------------------------------------------
#define BM 128
#define BN 64
#define BK 32
#define LDA 40   // bf16 elems/row (32 + 8 pad; 80B row keeps 16B segment alignment)
#define LDB 40

// ============================================================================
// Kernel 0: fp32 -> bf16 conversion (elementwise, vector4)
// ============================================================================
__global__ __launch_bounds__(256)
void cvt_kernel(const float* __restrict__ s, __bf16* __restrict__ d, int n4) {
  int i = blockIdx.x * 256 + threadIdx.x;
  if (i < n4) {
    v4f f = *(const v4f*)(s + 4*(size_t)i);
    *(v4bf*)(d + 4*(size_t)i) = cvt4(f);
  }
}

// ---- shared staging: double-buffered async tile copy ------------------------
// A tile: 128x32 bf16 (2 x 16B segs/thread), B tile: 64x32 bf16 (1 seg/thread)
__device__ __forceinline__ void stage_tiles(const __bf16* __restrict__ Ab,
                                            const __bf16* __restrict__ Bb,
                                            __bf16* As, __bf16* Bs,
                                            int t, int m0, int n0, int k0) {
  #pragma unroll
  for (int ss = 0; ss < 2; ++ss) {
    int seg = t + 256*ss;                 // 512 segs
    int row = seg >> 2, cg = seg & 3;
    async_cp16(Ab + (size_t)(m0 + row)*D_ + k0 + cg*8, &As[row*LDA + cg*8]);
  }
  {
    int row = t >> 2, cg = t & 3;         // 256 segs
    async_cp16(Bb + (size_t)(n0 + row)*D_ + k0 + cg*8, &Bs[row*LDB + cg*8]);
  }
}

// ============================================================================
// Kernel 1: qkv = xb @ Wqkvb^T  (M=16384, N=1536, K=512), bf16 in, bf16 out
// scatter: q(*0.125)[B,H,L,DH], k[B,H,L,DH], vT[B,H,DH,L]   (all bf16)
// ============================================================================
__global__ __launch_bounds__(256)
void qkv_gemm_kernel(const __bf16* __restrict__ X,   // [M, 512] bf16
                     const __bf16* __restrict__ W,   // [1536, 512] bf16
                     __bf16* __restrict__ qb, __bf16* __restrict__ kb,
                     __bf16* __restrict__ vtb)
{
  __shared__ __align__(16) __bf16 As[2 * BM * LDA];
  __shared__ __align__(16) __bf16 Bs[2 * BN * LDB];
  const int t = threadIdx.x;
  const int wave = t >> 5, lane = t & 31;
  const int lr = lane & 15, lh = lane >> 4;
  const int wm = wave >> 1, wn = wave & 1;          // 4x2 wave grid
  const int m0 = blockIdx.x * BM;
  const int n0 = blockIdx.y * BN;

  v8f acc[2][2] = {};
  int buf = 0;
  stage_tiles(X, W, As, Bs, t, m0, n0, 0);
  for (int k0 = 0; k0 < D_; k0 += BK) {
    if (k0 + BK < D_) {
      stage_tiles(X, W, As + (buf^1)*BM*LDA, Bs + (buf^1)*BN*LDB, t, m0, n0, k0 + BK);
      WAIT_ASYNC(0x3);                    // drain previous tile, keep next in flight
    } else {
      WAIT_ASYNC(0x0);
    }
    __syncthreads();
    const __bf16* asb = As + buf*BM*LDA;
    const __bf16* bsb = Bs + buf*BN*LDB;
    v16bf a[2], b[2];
    #pragma unroll
    for (int mf = 0; mf < 2; ++mf) {
      const __bf16* ap = &asb[(32*wm + 16*mf + lr)*LDA];
      a[mf] = cat8(*(const v8bf*)(ap + 16*lh), *(const v8bf*)(ap + 16*lh + 8));
    }
    #pragma unroll
    for (int nf = 0; nf < 2; ++nf)
      b[nf] = *(const v16bf*)(&bsb[(32*wn + 16*nf + lr)*LDB + 16*lh]);
    #pragma unroll
    for (int mf = 0; mf < 2; ++mf)
      #pragma unroll
      for (int nf = 0; nf < 2; ++nf)
        acc[mf][nf] = WMMA_BF16(a[mf], b[nf], acc[mf][nf]);
    __syncthreads();
    buf ^= 1;
  }
  // epilogue: D-fragment lane holds N = lr, rows M = r + 8*lh
  #pragma unroll
  for (int mf = 0; mf < 2; ++mf) {
    #pragma unroll
    for (int nf = 0; nf < 2; ++nf) {
      int n = n0 + 32*wn + 16*nf + lr;
      int which = n >> 9;                 // 0=q 1=k 2=v
      int d = n & 511, h = d >> 6, dh = d & 63;
      #pragma unroll
      for (int r = 0; r < 8; ++r) {
        int m = m0 + 32*wm + 16*mf + r + 8*lh;
        int l = m / B_, bb = m % B_;      // m = l*B + b  (x is [L,B,D])
        float v = acc[mf][nf][r];
        size_t bh = (size_t)(bb*H_ + h);
        if (which == 0)      qb [(bh*L_  + l )*DH_ + dh] = (__bf16)(v * 0.125f);
        else if (which == 1) kb [(bh*L_  + l )*DH_ + dh] = (__bf16)v;
        else                 vtb[(bh*DH_ + dh)*L_  + l ] = (__bf16)v;
      }
    }
  }
}

// ============================================================================
// Kernel 2: banded flash attention. One wave per 16-query tile. Pure bf16 loads.
// ============================================================================
__global__ __launch_bounds__(128)
void attn_kernel(const __bf16* __restrict__ qws, const __bf16* __restrict__ kws,
                 const __bf16* __restrict__ vtws, __bf16* __restrict__ aws)
{
  __shared__ __align__(16) __bf16 pLds[4][16*32];   // 1KB per wave
  const int t = threadIdx.x;
  const int wave = t >> 5, lane = t & 31;
  const int lr = lane & 15, lh = lane >> 4;
  const int bh = blockIdx.y;                        // b*H + h
  const int b = bh / H_, h = bh % H_;
  const int q0 = (blockIdx.x * 4 + wave) * 16;

  const __bf16* Q  = qws  + (size_t)bh * L_ * DH_;
  const __bf16* Kp = kws  + (size_t)bh * L_ * DH_;
  const __bf16* VT = vtws + (size_t)bh * DH_ * L_;

  // Q fragments (A layout); softmax scale already folded in at GEMM-1 epilogue
  v16bf qa[2];
  #pragma unroll
  for (int c = 0; c < 2; ++c) {
    const __bf16* qp = Q + (size_t)(q0 + lr)*DH_ + 32*c;
    qa[c] = cat8(*(const v8bf*)(qp + 16*lh), *(const v8bf*)(qp + 16*lh + 8));
  }

  float mrow[8], lrow[8];
  #pragma unroll
  for (int r = 0; r < 8; ++r) { mrow[r] = NEGF; lrow[r] = 0.f; }
  v8f o[4] = {};

  int js0  = (q0 >= 64) ? ((q0 - 64) & ~31) : 0;
  int jend = (q0 + 80 > L_) ? L_ : (q0 + 80);

  for (int j0 = js0; j0 < jend; j0 += 32) {
    // K^T B-fragments: [key tile jt][d chunk dc]
    v16bf kfr[2][2];
    #pragma unroll
    for (int jt = 0; jt < 2; ++jt) {
      const __bf16* kp = Kp + (size_t)(j0 + 16*jt + lr)*DH_;
      #pragma unroll
      for (int dc = 0; dc < 2; ++dc)
        kfr[jt][dc] = *(const v16bf*)(kp + 32*dc + 16*lh);
    }
    v8f s[2] = {};
    #pragma unroll
    for (int jt = 0; jt < 2; ++jt) {
      s[jt] = WMMA_BF16(qa[0], kfr[jt][0], s[jt]);
      s[jt] = WMMA_BF16(qa[1], kfr[jt][1], s[jt]);
    }
    // band mask + per-chunk row max
    float cmax[8];
    #pragma unroll
    for (int r = 0; r < 8; ++r) cmax[r] = NEGF;
    #pragma unroll
    for (int jt = 0; jt < 2; ++jt) {
      int j = j0 + 16*jt + lr;
      #pragma unroll
      for (int r = 0; r < 8; ++r) {
        int i = q0 + r + 8*lh;
        float v = s[jt][r];
        int dd = i - j; dd = dd < 0 ? -dd : dd;
        if (dd > PAD_) v = NEGF;
        s[jt][r] = v;
        cmax[r] = fmaxf(cmax[r], v);
      }
    }
    #pragma unroll
    for (int r = 0; r < 8; ++r) {                   // 16-lane row reduce (wave32)
      float v = cmax[r];
      v = fmaxf(v, __shfl_xor(v, 1, 32));
      v = fmaxf(v, __shfl_xor(v, 2, 32));
      v = fmaxf(v, __shfl_xor(v, 4, 32));
      v = fmaxf(v, __shfl_xor(v, 8, 32));
      cmax[r] = v;
    }
    float rsum[8];
    #pragma unroll
    for (int r = 0; r < 8; ++r) {
      float mnew = fmaxf(mrow[r], cmax[r]);
      float al = __expf(mrow[r] - mnew);
      mrow[r] = mnew; lrow[r] *= al;
      #pragma unroll
      for (int dt = 0; dt < 4; ++dt) o[dt][r] *= al;
      rsum[r] = 0.f;
    }
    // P -> LDS (D layout -> memory), bf16
    __bf16* pl = pLds[wave];
    #pragma unroll
    for (int jt = 0; jt < 2; ++jt)
      #pragma unroll
      for (int r = 0; r < 8; ++r) {
        float p = __expf(s[jt][r] - mrow[r]);
        rsum[r] += p;
        pl[(r + 8*lh)*32 + 16*jt + lr] = (__bf16)p;
      }
    #pragma unroll
    for (int r = 0; r < 8; ++r) {
      float v = rsum[r];
      v += __shfl_xor(v, 1, 32);
      v += __shfl_xor(v, 2, 32);
      v += __shfl_xor(v, 4, 32);
      v += __shfl_xor(v, 8, 32);
      lrow[r] += v;
    }
    // reload P as A fragment (wave-private LDS; explicit dscnt fence)
    asm volatile("s_wait_dscnt 0x0" ::: "memory");
    v16bf pa = cat8(*(const v8bf*)(pl + lr*32 + 16*lh),
                    *(const v8bf*)(pl + lr*32 + 16*lh + 8));
    // PV: B fragment from V^T rows (contiguous in j)
    #pragma unroll
    for (int dt = 0; dt < 4; ++dt) {
      const __bf16* vp = VT + (size_t)(16*dt + lr)*L_ + j0 + 16*lh;
      v16bf vb = *(const v16bf*)vp;
      o[dt] = WMMA_BF16(pa, vb, o[dt]);
    }
  }
  // normalize + store attn output as [L,B,D] bf16
  #pragma unroll
  for (int dt = 0; dt < 4; ++dt) {
    int dcol = h*DH_ + 16*dt + lr;
    #pragma unroll
    for (int r = 0; r < 8; ++r) {
      int li = q0 + r + 8*lh;
      aws[((size_t)li*B_ + b)*D_ + dcol] = (__bf16)(o[dt][r] / lrow[r]);
    }
  }
}

// ============================================================================
// Kernel 3: out = attn @ Wout^T + bout  (M=16384, N=512, K=512), fp32 out
// ============================================================================
__global__ __launch_bounds__(256)
void out_gemm_kernel(const __bf16* __restrict__ X,   // [M, 512] bf16
                     const __bf16* __restrict__ W,   // [512, 512] bf16
                     const float* __restrict__ bias,
                     float* __restrict__ out)
{
  __shared__ __align__(16) __bf16 As[2 * BM * LDA];
  __shared__ __align__(16) __bf16 Bs[2 * BN * LDB];
  const int t = threadIdx.x;
  const int wave = t >> 5, lane = t & 31;
  const int lr = lane & 15, lh = lane >> 4;
  const int wm = wave >> 1, wn = wave & 1;
  const int m0 = blockIdx.x * BM;
  const int n0 = blockIdx.y * BN;

  v8f acc[2][2] = {};
  int buf = 0;
  stage_tiles(X, W, As, Bs, t, m0, n0, 0);
  for (int k0 = 0; k0 < D_; k0 += BK) {
    if (k0 + BK < D_) {
      stage_tiles(X, W, As + (buf^1)*BM*LDA, Bs + (buf^1)*BN*LDB, t, m0, n0, k0 + BK);
      WAIT_ASYNC(0x3);
    } else {
      WAIT_ASYNC(0x0);
    }
    __syncthreads();
    const __bf16* asb = As + buf*BM*LDA;
    const __bf16* bsb = Bs + buf*BN*LDB;
    v16bf a[2], b[2];
    #pragma unroll
    for (int mf = 0; mf < 2; ++mf) {
      const __bf16* ap = &asb[(32*wm + 16*mf + lr)*LDA];
      a[mf] = cat8(*(const v8bf*)(ap + 16*lh), *(const v8bf*)(ap + 16*lh + 8));
    }
    #pragma unroll
    for (int nf = 0; nf < 2; ++nf)
      b[nf] = *(const v16bf*)(&bsb[(32*wn + 16*nf + lr)*LDB + 16*lh]);
    #pragma unroll
    for (int mf = 0; mf < 2; ++mf)
      #pragma unroll
      for (int nf = 0; nf < 2; ++nf)
        acc[mf][nf] = WMMA_BF16(a[mf], b[nf], acc[mf][nf]);
    __syncthreads();
    buf ^= 1;
  }
  #pragma unroll
  for (int mf = 0; mf < 2; ++mf) {
    #pragma unroll
    for (int nf = 0; nf < 2; ++nf) {
      int n = n0 + 32*wn + 16*nf + lr;
      float bv = bias[n];
      #pragma unroll
      for (int r = 0; r < 8; ++r) {
        int m = m0 + 32*wm + 16*mf + r + 8*lh;
        out[(size_t)m*D_ + n] = acc[mf][nf][r] + bv;
      }
    }
  }
}

// ============================================================================
extern "C" void kernel_launch(void* const* d_in, const int* in_sizes, int n_in,
                              void* d_out, int out_size, void* d_ws, size_t ws_size,
                              hipStream_t stream) {
  const float* x    = (const float*)d_in[0];   // [L,B,D]
  const float* Wqkv = (const float*)d_in[1];   // [3D,D]
  const float* Wout = (const float*)d_in[2];   // [D,D]
  const float* bout = (const float*)d_in[3];   // [D]
  float* out = (float*)d_out;                  // [L,B,D]

  const size_t QK = (size_t)B_ * H_ * L_ * DH_;    // 8.39M elems
  __bf16* ws    = (__bf16*)d_ws;
  __bf16* xb    = ws;                              // [M,512]
  __bf16* wqkvb = xb    + (size_t)M_ * D_;         // [1536,512]
  __bf16* woutb = wqkvb + (size_t)3 * D_ * D_;     // [512,512]
  __bf16* qb    = woutb + (size_t)D_ * D_;         // [B,H,L,DH] (pre-scaled)
  __bf16* kbuf  = qb    + QK;                      // [B,H,L,DH]
  __bf16* vtb   = kbuf  + QK;                      // [B,H,DH,L]
  __bf16* ab    = vtb   + QK;                      // [L,B,D]     (~87MB total)

  // fp32 -> bf16 conversion passes (all counts divisible by 1024)
  cvt_kernel<<<(M_*D_)/1024,        256, 0, stream>>>(x,    xb,    (M_*D_)/4);
  cvt_kernel<<<(3*D_*D_)/1024,      256, 0, stream>>>(Wqkv, wqkvb, (3*D_*D_)/4);
  cvt_kernel<<<(D_*D_)/1024,        256, 0, stream>>>(Wout, woutb, (D_*D_)/4);

  dim3 g1(M_ / BM, (3 * D_) / BN);    // (128, 24)
  qkv_gemm_kernel<<<g1, 256, 0, stream>>>(xb, wqkvb, qb, kbuf, vtb);

  dim3 g2(L_ / (16 * 4), B_ * H_);    // (32, 64)
  attn_kernel<<<g2, 128, 0, stream>>>(qb, kbuf, vtb, ab);

  dim3 g3(M_ / BM, D_ / BN);          // (128, 8)
  out_gemm_kernel<<<g3, 256, 0, stream>>>(ab, woutb, bout, out);
}
